// MultiHeadAttention_51694226374990
// MI455X (gfx1250) — matrix operand order repaired and checked by reading, not compiled, not run on hardware
//
#include <hip/hip_runtime.h>
#include <hip/hip_bf16.h>
#include <math.h>

// MI455X / gfx1250, wave32, WMMA (not MFMA).
// MHA: B=2, S=4096, D=512, H=8, hd=64.  fp32 in/out, bf16 WMMA internally.

#define DMODEL 512
#define SEQ    4096
#define BATCH  2
#define NHEADS 8
#define HEADD  64

// LDS row padding: +8 ushorts (16B). Row stride 144B => bank step 36,
// gcd(36,64)=4 => 16 fragment lanes hit 16 distinct banks; 16B alignment kept.
#define KPAD 8

typedef __attribute__((ext_vector_type(16))) __bf16 bf16x16;
typedef __attribute__((ext_vector_type(4)))  __bf16 bf16x4;
typedef __attribute__((ext_vector_type(8)))  float  f32x8;
typedef __attribute__((ext_vector_type(4)))  float  f32x4v;
typedef __attribute__((ext_vector_type(4)))  int    i32x4;

// Address-space-qualified int4 for the gfx1250 async-copy builtin:
// param0 = global (addrspace(1)) int4*, param1 = LDS (addrspace(3)) int4*.
typedef __attribute__((address_space(1))) i32x4 gbl_i32x4;
typedef __attribute__((address_space(3))) i32x4 lds_i32x4;

union Frag16 {
  bf16x16        v;
  unsigned int   u32[8];
  unsigned short u16[16];
};

// fp32 -> bf16 via native conversion (backend picks v_cvt_pk_bf16_f32 if present)
__device__ __forceinline__ unsigned short f2bf(float f) {
  __bf16 h = (__bf16)f;
  return __builtin_bit_cast(unsigned short, h);
}

__device__ __forceinline__ unsigned long long pack4bf(float4 f) {
  f32x4v fv = {f.x, f.y, f.z, f.w};
  bf16x4 h = __builtin_convertvector(fv, bf16x4);
  return __builtin_bit_cast(unsigned long long, h);
}

// ---- CDNA5 async copy global -> LDS (ASYNCcnt), guarded ----------------------
#if defined(__AMDGCN__) && __has_builtin(__builtin_amdgcn_global_load_async_to_lds_b128)
#define HAVE_ASYNC_LDS 1
#else
#define HAVE_ASYNC_LDS 0
#endif

__device__ __forceinline__ void async_copy_b128(const unsigned short* g,
                                                unsigned short* l) {
#if HAVE_ASYNC_LDS
  __builtin_amdgcn_global_load_async_to_lds_b128(
      (gbl_i32x4*)g, (lds_i32x4*)l, 0, 0);
#else
  *(uint4*)l = *(const uint4*)g;
#endif
}

__device__ __forceinline__ void async_wait0() {
#if HAVE_ASYNC_LDS
#if __has_builtin(__builtin_amdgcn_s_wait_asynccnt)
  __builtin_amdgcn_s_wait_asynccnt(0);
#else
  asm volatile("s_wait_asynccnt 0x0" ::: "memory");
#endif
#endif
}

// ---------------------------------------------------------------------------
// Kernel 1: fused QKV projection.  y = x @ W^T + b, bf16 output.
//   Q: [B,H,S,hd] scaled by 1/sqrt(hd)   K: [B,H,S,hd]   V: [B,H,hd,S]
// (V head-transposed so attention can async-DMA it straight into LDS.)
// Block: 256 threads (8 waves); tile 256 rows x 64 cols; 32 rows per wave.
// ---------------------------------------------------------------------------
__global__ __launch_bounds__(256) void qkv_proj_wmma(
    const float* __restrict__ x,
    const float* __restrict__ Wq, const float* __restrict__ bq,
    const float* __restrict__ Wk, const float* __restrict__ bk,
    const float* __restrict__ Wv, const float* __restrict__ bv,
    unsigned short* __restrict__ Qo, unsigned short* __restrict__ Ko,
    unsigned short* __restrict__ Vo)
{
  const int z = blockIdx.z;
  const float* W    = (z == 0) ? Wq : (z == 1) ? Wk : Wv;
  const float* bias = (z == 0) ? bq : (z == 1) ? bk : bv;
  unsigned short* __restrict__ out = (z == 0) ? Qo : (z == 1) ? Ko : Vo;
  const bool  vtrans = (z == 2);            // V stored [B,H,hd,S]
  const float oscale = (z == 0) ? 0.125f : 1.0f;  // fold 1/sqrt(hd) into Q

  __shared__ unsigned short Xs[256][32 + KPAD];  // x tile, bf16, [row][k]
  __shared__ unsigned short Ws[64][32 + KPAD];   // W tile, bf16, [n][k]

  const int t    = threadIdx.x;
  const int wave = t >> 5;
  const int lane = t & 31;
  const int lm   = lane & 15;
  const int hi   = lane >> 4;               // lane-half select
  const int m0   = blockIdx.x * 256;        // row tile over B*S = 8192
  const int n0   = blockIdx.y * 64;         // col tile over D = 512

  f32x8 acc[2][4] = {};                     // 32x64 output per wave

  #pragma unroll 1
  for (int kc = 0; kc < DMODEL; kc += 32) {
    // stage x: 256x32 f32 -> bf16 (float4 loads, 64-bit packed LDS stores)
    for (int i = t; i < 256 * 8; i += 256) {
      int row = i >> 3, c4 = (i & 7) * 4;
      float4 f = *(const float4*)&x[(size_t)(m0 + row) * DMODEL + kc + c4];
      *(unsigned long long*)&Xs[row][c4] = pack4bf(f);
    }
    // stage W: 64x32 f32 -> bf16
    for (int i = t; i < 64 * 8; i += 256) {
      int row = i >> 3, c4 = (i & 7) * 4;
      float4 f = *(const float4*)&W[(size_t)(n0 + row) * DMODEL + kc + c4];
      *(unsigned long long*)&Ws[row][c4] = pack4bf(f);
    }
    __syncthreads();

    // A fragments (16x32 bf16): lane holds row M=lane%16; Kbase 0 / 16 by half
    Frag16 a[2];
    #pragma unroll
    for (int mi = 0; mi < 2; ++mi) {
      const unsigned int* p =
          (const unsigned int*)&Xs[32 * wave + 16 * mi + lm][0];
      const int kb2 = hi * 8;
      #pragma unroll
      for (int j = 0; j < 8; ++j) {
        int ku = kb2 + j;  // K = kbase + {0..15}
        a[mi].u32[j] = p[ku];
      }
    }

    #pragma unroll
    for (int nf = 0; nf < 4; ++nf) {
      // B fragment (32x16): lane holds col N=lane%16; K = hi*16 + e contiguous
      Frag16 bfrag;
      const unsigned int* p = (const unsigned int*)&Ws[16 * nf + lm][hi * 16];
      #pragma unroll
      for (int j = 0; j < 8; ++j) bfrag.u32[j] = p[j];
      acc[0][nf] = __builtin_amdgcn_wmma_f32_16x16x32_bf16(
          false, a[0].v, false, bfrag.v, (short)0, acc[0][nf], false, false);
      acc[1][nf] = __builtin_amdgcn_wmma_f32_16x16x32_bf16(
          false, a[1].v, false, bfrag.v, (short)0, acc[1][nf], false, false);
    }
    __syncthreads();
  }

  // Epilogue: +bias, optional Q-scale, bf16 store in head-major layout
  #pragma unroll
  for (int nf = 0; nf < 4; ++nf) {
    const int n    = n0 + 16 * nf + lm;
    const float bb = bias[n];
    const int head = n >> 6, d = n & 63;
    #pragma unroll
    for (int mi = 0; mi < 2; ++mi) {
      #pragma unroll
      for (int r = 0; r < 8; ++r) {
        int gm   = m0 + 32 * wave + 16 * mi + r + 8 * hi;  // C: M = r + 8*half
        int bidx = gm >> 12;                               // / SEQ
        int s    = gm & (SEQ - 1);
        size_t o = vtrans
            ? (((size_t)bidx * NHEADS + head) * HEADD + d) * SEQ + s
            : (((size_t)bidx * NHEADS + head) * SEQ + s) * HEADD + d;
        out[o] = f2bf((acc[mi][nf][r] + bb) * oscale);
      }
    }
  }
}

// ---------------------------------------------------------------------------
// Kernel 2: flash attention.  Block = 8 waves x 32 query rows = 256 rows of
// one (b,h).  K and V chunks double-buffered in LDS and filled by async DMA
// (ASYNCcnt) one chunk ahead of compute; Q fragments resident in VGPRs.
// ---------------------------------------------------------------------------
__global__ __launch_bounds__(256) void attn_wmma(
    const unsigned short* __restrict__ Qb,
    const unsigned short* __restrict__ Kb,   // [B,H,S,hd]
    const unsigned short* __restrict__ Vb,   // [B,H,hd,S]
    float* __restrict__ out)
{
  __shared__ unsigned short Ks[2][64][64 + KPAD];  // K chunks [key][hd]
  __shared__ unsigned short Vt[2][64][64 + KPAD];  // V chunks [hd][key]
  __shared__ unsigned short Ps[8][32][64 + KPAD];  // per-wave P tiles

  const int t    = threadIdx.x;
  const int wave = t >> 5;
  const int lane = t & 31;
  const int lm   = lane & 15;
  const int hi   = lane >> 4;
  const int b    = blockIdx.z;
  const int h    = blockIdx.y;
  const int sw   = blockIdx.x * 256 + 32 * wave;   // first query row of wave
  const size_t base = (((size_t)b * NHEADS + h) * SEQ) * HEADD;  // == H-major base

  // Q A-fragments: two 16-row M-tiles x two 32-wide K-chunks, resident.
  // (Q already scaled by 1/sqrt(hd) at projection time.)
  Frag16 qa[2][2];
  {
    const int kb = hi * 16;
    #pragma unroll
    for (int mi = 0; mi < 2; ++mi) {
      const unsigned short* qp = Qb + base + (size_t)(sw + 16 * mi + lm) * HEADD;
      #pragma unroll
      for (int kf = 0; kf < 2; ++kf) {
        #pragma unroll
        for (int j = 0; j < 8; ++j) {
          int k = 32 * kf + kb + 2 * j;
          qa[mi][kf].u32[j] = *(const unsigned int*)(qp + k);
        }
      }
    }
  }

  float m_run[2][8], l_run[2][8];
  f32x8 o_acc[2][4] = {};
  #pragma unroll
  for (int mi = 0; mi < 2; ++mi)
    #pragma unroll
    for (int r = 0; r < 8; ++r) { m_run[mi][r] = -3.0e38f; l_run[mi][r] = 0.0f; }

  // Prologue: async-fill chunk 0 into buffer 0 (K rows by key, V rows by hd)
  for (int u = t; u < 512; u += 256) {
    int row = u >> 3, cq = (u & 7) * 8;
    async_copy_b128(Kb + base + (size_t)row * HEADD + cq, &Ks[0][row][cq]);
    async_copy_b128(Vb + base + (size_t)row * SEQ + cq,  &Vt[0][row][cq]);
  }

  #pragma unroll 1
  for (int jc = 0; jc < SEQ; jc += 64) {
    const int buf = (jc >> 6) & 1;

    async_wait0();      // this wave's chunk-jc copies done
    __syncthreads();    // => everyone's copies done; prev reads of buf^1 done

    // Issue next chunk into the other buffer: DMA overlaps this chunk's WMMAs
    if (jc + 64 < SEQ) {
      const int jn = jc + 64;
      for (int u = t; u < 512; u += 256) {
        int row = u >> 3, cq = (u & 7) * 8;
        async_copy_b128(Kb + base + (size_t)(jn + row) * HEADD + cq,
                        &Ks[buf ^ 1][row][cq]);
        async_copy_b128(Vb + base + (size_t)row * SEQ + jn + cq,
                        &Vt[buf ^ 1][row][cq]);
      }
      // L2 prefetch two chunks ahead (global_prefetch_b8)
      if (jc + 128 < SEQ) {
        __builtin_prefetch(Kb + base + (size_t)(jc + 128 + (t >> 2)) * HEADD +
                               (t & 3) * 16, 0, 0);
        __builtin_prefetch(Vb + base + (size_t)(t >> 2) * SEQ + jc + 128 +
                               (t & 3) * 16, 0, 0);
      }
    }

    // ---- scores: S = Q @ K^T  (32 x 64), 16 WMMAs, B-frags reused x2 ----
    f32x8 sc[2][4] = {};
    #pragma unroll
    for (int kf = 0; kf < 2; ++kf) {
      #pragma unroll
      for (int nf = 0; nf < 4; ++nf) {
        Frag16 bk;  // contraction = hd, N = key column
        const unsigned int* p =
            (const unsigned int*)&Ks[buf][16 * nf + lm][32 * kf + hi * 16];
        #pragma unroll
        for (int j = 0; j < 8; ++j) bk.u32[j] = p[j];
        sc[0][nf] = __builtin_amdgcn_wmma_f32_16x16x32_bf16(
            false, qa[0][kf].v, false, bk.v, (short)0, sc[0][nf], false, false);
        sc[1][nf] = __builtin_amdgcn_wmma_f32_16x16x32_bf16(
            false, qa[1][kf].v, false, bk.v, (short)0, sc[1][nf], false, false);
      }
    }

    // ---- online softmax (fp32); rows live in 16-lane halves ----
    #pragma unroll
    for (int mi = 0; mi < 2; ++mi) {
      #pragma unroll
      for (int r = 0; r < 8; ++r) {
        float s0 = sc[mi][0][r], s1 = sc[mi][1][r];
        float s2 = sc[mi][2][r], s3 = sc[mi][3][r];
        float v = fmaxf(fmaxf(s0, s1), fmaxf(s2, s3));
        #pragma unroll
        for (int off = 8; off >= 1; off >>= 1)
          v = fmaxf(v, __shfl_xor(v, off, 32));
        float mnew  = fmaxf(m_run[mi][r], v);
        float alpha = __expf(m_run[mi][r] - mnew);
        m_run[mi][r] = mnew;
        float e0 = __expf(s0 - mnew), e1 = __expf(s1 - mnew);
        float e2 = __expf(s2 - mnew), e3 = __expf(s3 - mnew);
        int prow = 16 * mi + r + 8 * hi;
        Ps[wave][prow][ 0 + lm] = f2bf(e0);
        Ps[wave][prow][16 + lm] = f2bf(e1);
        Ps[wave][prow][32 + lm] = f2bf(e2);
        Ps[wave][prow][48 + lm] = f2bf(e3);
        float rs = e0 + e1 + e2 + e3;
        #pragma unroll
        for (int off = 8; off >= 1; off >>= 1)
          rs += __shfl_xor(rs, off, 32);
        l_run[mi][r] = l_run[mi][r] * alpha + rs;
        o_acc[mi][0][r] *= alpha; o_acc[mi][1][r] *= alpha;
        o_acc[mi][2][r] *= alpha; o_acc[mi][3][r] *= alpha;
      }
    }

    // wave-local LDS RAW fence before re-reading Ps in A layout
    asm volatile("s_wait_dscnt 0x0" ::: "memory");
    __builtin_amdgcn_wave_barrier();

    // P re-swizzled into A fragments
    Frag16 pa[2][2];
    {
      const int kb2 = hi * 8;
      #pragma unroll
      for (int mi = 0; mi < 2; ++mi) {
        #pragma unroll
        for (int kf = 0; kf < 2; ++kf) {
          const unsigned int* p =
              (const unsigned int*)&Ps[wave][16 * mi + lm][32 * kf];
          #pragma unroll
          for (int j = 0; j < 8; ++j) {
            int ku = kb2 + j;
            pa[mi][kf].u32[j] = p[ku];
          }
        }
      }
    }

    // ---- O += P @ V  (32 x 64), 16 WMMAs, B-frags reused x2 ----
    #pragma unroll
    for (int kf = 0; kf < 2; ++kf) {
      #pragma unroll
      for (int of = 0; of < 4; ++of) {
        Frag16 bvf;  // contraction = key, N = hd column (Vt is [hd][key])
        const unsigned int* p =
            (const unsigned int*)&Vt[buf][16 * of + lm][32 * kf + hi * 16];
        #pragma unroll
        for (int j = 0; j < 8; ++j) bvf.u32[j] = p[j];
        o_acc[0][of] = __builtin_amdgcn_wmma_f32_16x16x32_bf16(
            false, pa[0][kf].v, false, bvf.v, (short)0, o_acc[0][of], false, false);
        o_acc[1][of] = __builtin_amdgcn_wmma_f32_16x16x32_bf16(
            false, pa[1][kf].v, false, bvf.v, (short)0, o_acc[1][of], false, false);
      }
    }
  }

  // Epilogue: normalize and write fp32 [B,S,D]
  const size_t ob = (size_t)b * SEQ * DMODEL + (size_t)h * HEADD;
  #pragma unroll
  for (int mi = 0; mi < 2; ++mi) {
    #pragma unroll
    for (int r = 0; r < 8; ++r) {
      float inv = 1.0f / l_run[mi][r];
      int s = sw + 16 * mi + r + 8 * hi;
      float* op = out + ob + (size_t)s * DMODEL + lm;
      #pragma unroll
      for (int of = 0; of < 4; ++of)
        op[16 * of] = o_acc[mi][of][r] * inv;
    }
  }
}

// ---------------------------------------------------------------------------
extern "C" void kernel_launch(void* const* d_in, const int* in_sizes, int n_in,
                              void* d_out, int out_size, void* d_ws, size_t ws_size,
                              hipStream_t stream) {
  (void)in_sizes; (void)n_in; (void)out_size; (void)ws_size;

  const float* x  = (const float*)d_in[0];
  const float* Wq = (const float*)d_in[1];
  const float* bq = (const float*)d_in[2];
  const float* Wk = (const float*)d_in[3];
  const float* bk = (const float*)d_in[4];
  const float* Wv = (const float*)d_in[5];
  const float* bv = (const float*)d_in[6];
  float* out = (float*)d_out;

  const size_t elems = (size_t)BATCH * SEQ * DMODEL;   // 4M -> 8 MB bf16 each
  unsigned short* Qb = (unsigned short*)d_ws;
  unsigned short* Kb = Qb + elems;
  unsigned short* Vb = Kb + elems;                     // 24 MB workspace total

  dim3 g1(BATCH * SEQ / 256, DMODEL / 64, 3);          // 32 x 8 x 3 blocks
  qkv_proj_wmma<<<g1, 256, 0, stream>>>(x, Wq, bq, Wk, bk, Wv, bv, Qb, Kb, Vb);

  dim3 g2(SEQ / 256, NHEADS, BATCH);                   // 16 x 8 x 2 blocks
  attn_wmma<<<g2, 256, 0, stream>>>(Qb, Kb, Vb, out);
}
